// OuterProductMean_14181982011859
// MI455X (gfx1250) — matrix operand order repaired and checked by reading, not compiled, not run on hardware
//
#include <hip/hip_runtime.h>

// ---------------------------------------------------------------------------
// OuterProductMean: out[i,j,h] = (1/S) * sum_s a[s,i,h] * b[s,j,h]
// 64 independent GEMMs (one per h), M=N=512, K=S=1024, fp32 in/out.
// Strategy: bf16 hi/lo split-GEMM on v_wmma_f32_16x16x32_bf16 (3 WMMAs per
// tile-step -> fp32-class accuracy at ~3x bf16 cost),
// LDS staging with on-the-fly f32->bf16 split using the
// native f32->bf16 converter (v_cvt_pk_bf16_f32) instead of bit math.
// ---------------------------------------------------------------------------

typedef __attribute__((ext_vector_type(16))) __bf16    bf16x16;
typedef __attribute__((ext_vector_type(2)))  __bf16    bf16x2;
typedef __attribute__((ext_vector_type(8)))  float     v8f;
typedef __attribute__((ext_vector_type(4)))  float     f32x4;
typedef __attribute__((ext_vector_type(4)))  unsigned  u32x4;

#define S_DIM 1024
#define I_DIM 512
#define H_DIM 64

#define TI 64      // i-tile per block
#define TJ 64      // j-tile per block
#define HC 4       // h planes per block
#define KC 32      // s-chunk (WMMA K)

#define ROWB 80    // bytes per k-row in LDS: 32*2B + 16B pad (bank spread)
#define PLANE_BYTES (HC * TI * ROWB)   // 20480
#define OFF_AHI 0
#define OFF_ALO (PLANE_BYTES * 1)
#define OFF_BHI (PLANE_BYTES * 2)
#define OFF_BLO (PLANE_BYTES * 3)
#define LDS_BYTES (PLANE_BYTES * 4)    // 81920

struct Pref { f32x4 x0, x1; };

// Load the (2*kp, row) and (2*kp+1, row) h-quads for one input tensor.
__device__ __forceinline__ Pref load_pair(const float* __restrict__ p,
                                          int s0, int kp, int row, int h0) {
    const int base = ((s0 + 2 * kp) * I_DIM + row) * H_DIM + h0;
    Pref r;
    r.x0 = *reinterpret_cast<const f32x4*>(p + base);
    r.x1 = *reinterpret_cast<const f32x4*>(p + base + I_DIM * H_DIM);
    return r;
}

// Split to bf16 hi/lo (hi = RNE truncation, lo = RNE(residual)), pack k-pairs,
// store b32 into [h][row][k] planes. fptrunc -> v_cvt_pk_bf16_f32,
// residual -> v_fma_mix_f32_bf16 (verified pattern-match in round 1).
__device__ __forceinline__ void cvt_store(char* smem, int off_hi, int off_lo,
                                          int kp, int row, Pref r) {
#pragma unroll
    for (int h = 0; h < HC; ++h) {
        const float f0 = r.x0[h], f1 = r.x1[h];
        bf16x2 hi2;
        hi2.x = (__bf16)f0;
        hi2.y = (__bf16)f1;
        const float r0 = f0 - (float)hi2.x;   // exact residual
        const float r1 = f1 - (float)hi2.y;
        bf16x2 lo2;
        lo2.x = (__bf16)r0;
        lo2.y = (__bf16)r1;
        const int addr = (h * TI + row) * ROWB + kp * 4;
        *reinterpret_cast<bf16x2*>(smem + off_hi + addr) = hi2;
        *reinterpret_cast<bf16x2*>(smem + off_lo + addr) = lo2;
    }
}

union FragU { u32x4 q[2]; bf16x16 v; };

// A fragment (16x32 bf16): lane m=lane&15, g=lane>>4.
// VGPR0-3: k = 8g + 0..7 (16B), VGPR4-7: k = 16 + 8g + 0..7 (16B @ +32B).
__device__ __forceinline__ bf16x16 load_fragA(const char* smem, int off,
                                              int h, int rbase, int m, int g) {
    const int addr = off + (h * TI + rbase + m) * ROWB + 16 * g;
    FragU u;
    u.q[0] = *reinterpret_cast<const u32x4*>(smem + addr);
    u.q[1] = *reinterpret_cast<const u32x4*>(smem + addr + 32);
    return u.v;
}

// B fragment (32x16 bf16): lane n=lane&15, g=lane>>4.
// VGPR0-3: k = 8g + 0..7 (16B), VGPR4-7: k = 16 + 8g + 0..7 (16B @ +32B).
__device__ __forceinline__ bf16x16 load_fragB(const char* smem, int off,
                                              int h, int rbase, int n, int g) {
    const int addr = off + (h * TI + rbase + n) * ROWB + 16 * g;
    FragU u;
    u.q[0] = *reinterpret_cast<const u32x4*>(smem + addr);
    u.q[1] = *reinterpret_cast<const u32x4*>(smem + addr + 32);
    return u.v;
}

__device__ __forceinline__ v8f wmma_bf16(bf16x16 a, bf16x16 b, v8f c) {
    return __builtin_amdgcn_wmma_f32_16x16x32_bf16(
        /*neg_a=*/false, a, /*neg_b=*/false, b,
        /*c_mod=*/(short)0, c, /*reuse_a=*/false, /*reuse_b=*/false);
}

__global__ __launch_bounds__(512)
void opm_wmma_split_kernel(const float* __restrict__ A,
                           const float* __restrict__ B,
                           float* __restrict__ Out) {
    __shared__ __align__(16) char smem[LDS_BYTES];

    const int t    = threadIdx.x;
    const int lane = t & 31;
    const int wave = t >> 5;

    const int i0 = blockIdx.x * TI;
    const int j0 = blockIdx.y * TJ;
    const int h0 = blockIdx.z * HC;

    // wave -> (h plane, 2x2 quadrant of the 4x4 WMMA-tile grid)
    const int wh   = wave >> 2;          // 0..3
    const int quad = wave & 3;
    const int mi0  = (quad & 1) * 2;     // tile row base (0 or 2)
    const int nj0  = (quad >> 1) * 2;    // tile col base (0 or 2)

    const int lm = lane & 15;            // A-frag m / B-frag n / C col n
    const int g  = lane >> 4;

    // staging work items: row = t&63, k-pairs {t>>6, (t>>6)+8}
    const int srow = t & 63;
    const int skp  = t >> 6;             // 0..7

    v8f acc[2][2];
#pragma unroll
    for (int r = 0; r < 2; ++r)
#pragma unroll
        for (int c = 0; c < 2; ++c)
            acc[r][c] = (v8f){0.f, 0.f, 0.f, 0.f, 0.f, 0.f, 0.f, 0.f};

    // prefetch chunk 0
    Pref pa0 = load_pair(A, 0, skp,     i0 + srow, h0);
    Pref pa1 = load_pair(A, 0, skp + 8, i0 + srow, h0);
    Pref pb0 = load_pair(B, 0, skp,     j0 + srow, h0);
    Pref pb1 = load_pair(B, 0, skp + 8, j0 + srow, h0);

    for (int s0 = 0; s0 < S_DIM; s0 += KC) {
        __syncthreads();   // previous iteration finished reading LDS
        cvt_store(smem, OFF_AHI, OFF_ALO, skp,     srow, pa0);
        cvt_store(smem, OFF_AHI, OFF_ALO, skp + 8, srow, pa1);
        cvt_store(smem, OFF_BHI, OFF_BLO, skp,     srow, pb0);
        cvt_store(smem, OFF_BHI, OFF_BLO, skp + 8, srow, pb1);

        const int s1 = s0 + KC;
        if (s1 < S_DIM) {  // prefetch next chunk while this one computes
            pa0 = load_pair(A, s1, skp,     i0 + srow, h0);
            pa1 = load_pair(A, s1, skp + 8, i0 + srow, h0);
            pb0 = load_pair(B, s1, skp,     j0 + srow, h0);
            pb1 = load_pair(B, s1, skp + 8, j0 + srow, h0);
        }
        __syncthreads();   // staged data visible

        bf16x16 aH[2], aL[2], bH[2], bL[2];
#pragma unroll
        for (int r = 0; r < 2; ++r) {
            aH[r] = load_fragA(smem, OFF_AHI, wh, (mi0 + r) * 16, lm, g);
            aL[r] = load_fragA(smem, OFF_ALO, wh, (mi0 + r) * 16, lm, g);
        }
#pragma unroll
        for (int c = 0; c < 2; ++c) {
            bH[c] = load_fragB(smem, OFF_BHI, wh, (nj0 + c) * 16, lm, g);
            bL[c] = load_fragB(smem, OFF_BLO, wh, (nj0 + c) * 16, lm, g);
        }

#pragma unroll
        for (int r = 0; r < 2; ++r)
#pragma unroll
            for (int c = 0; c < 2; ++c) {
                acc[r][c] = wmma_bf16(aH[r], bH[c], acc[r][c]); // hi*hi
                acc[r][c] = wmma_bf16(aH[r], bL[c], acc[r][c]); // hi*lo
                acc[r][c] = wmma_bf16(aL[r], bH[c], acc[r][c]); // lo*hi
            }
    }

    // ---- epilogue: transpose through LDS for 16B-coalesced global stores ----
    __syncthreads();
    // outbuf[i][j][h]: row stride 260 floats (64*4 + 4 pad) = 1040 B
    float* ob = reinterpret_cast<float*>(smem);
    const float scale = 1.0f / (float)S_DIM;
#pragma unroll
    for (int r = 0; r < 2; ++r)
#pragma unroll
        for (int c = 0; c < 2; ++c) {
#pragma unroll
            for (int v = 0; v < 8; ++v) {
                const int mm = (mi0 + r) * 16 + v + 8 * g;   // C/D: M = v + 8g
                const int nn = (nj0 + c) * 16 + lm;          // C/D: N = lane&15
                ob[mm * 260 + nn * 4 + wh] = acc[r][c][v] * scale;
            }
        }
    __syncthreads();

#pragma unroll
    for (int p = 0; p < 8; ++p) {
        const int q  = t + p * 512;      // 0..4095
        const int jb = q & 63;
        const int ib = q >> 6;
        f32x4 val = *reinterpret_cast<const f32x4*>(smem + ib * 1040 + jb * 16);
        const int oidx = ((i0 + ib) * I_DIM + (j0 + jb)) * H_DIM + h0;
        *reinterpret_cast<f32x4*>(Out + oidx) = val;
    }
}

extern "C" void kernel_launch(void* const* d_in, const int* in_sizes, int n_in,
                              void* d_out, int out_size, void* d_ws, size_t ws_size,
                              hipStream_t stream) {
    (void)in_sizes; (void)n_in; (void)out_size; (void)d_ws; (void)ws_size;
    const float* a = (const float*)d_in[0];
    const float* b = (const float*)d_in[1];
    float* out = (float*)d_out;

    dim3 grid(I_DIM / TI, I_DIM / TJ, H_DIM / HC);   // 8 x 8 x 16
    dim3 block(512);                                  // 16 waves (wave32)
    opm_wmma_split_kernel<<<grid, block, 0, stream>>>(a, b, out);
}
